// FusedProjectionPlusCrossEntropyLoss_33827162423620
// MI455X (gfx1250) — matrix operand order repaired and checked by reading, not compiled, not run on hardware
//
#include <hip/hip_runtime.h>
#include <hip/hip_bf16.h>

#define DIM   4096
#define NTOK  4096
#define NCLS  50257
#define BM    128
#define BN    128
#define BK    64
#define CB    ((NCLS + BN - 1) / BN)   /* 393 class blocks */
#define KT    (DIM / BK)               /* 64 K stages */

#define ASTR  88    /* bf16 elems per LDS row: 64 + 24 pad -> 176B, 16B aligned, conflict-free */
#define BSTR  88
#define LSTR  68    /* float stride of logits scratch rows */

typedef __attribute__((ext_vector_type(16))) __bf16 v16bf;
typedef __attribute__((ext_vector_type(8)))  float  v8f;

union FragU { uint4 q[2]; v16bf v; };

__device__ __forceinline__ unsigned int pk_bf16(float a, float b) {
    // round-to-nearest-even fp32 -> bf16, packed pair
    unsigned int ua = __float_as_uint(a);
    unsigned int ub = __float_as_uint(b);
    ua = (ua + 0x7fffu + ((ua >> 16) & 1u)) >> 16;
    ub = (ub + 0x7fffu + ((ub >> 16) & 1u)) >> 16;
    return ua | (ub << 16);
}

// Stage 1: one (row-block, class-block) tile: bf16 WMMA GEMM -> online
// per-row (max, sum-exp) partials + target-logit capture.
__global__ __launch_bounds__(256) void ce_partial_kernel(
    const float* __restrict__ x, const int* __restrict__ targ,
    const float* __restrict__ W, float* __restrict__ Mpart,
    float* __restrict__ Spart, float* __restrict__ Tlog)
{
    __shared__ alignas(16) unsigned char smem[BM * ASTR * 2 + BN * BSTR * 2]; // 45056 B
    __bf16* As = (__bf16*)smem;                    // BM x ASTR
    __bf16* Bs = (__bf16*)(smem + BM * ASTR * 2);  // BN x BSTR
    float*  Ls = (float*)smem;                     // reused after GEMM: BM x LSTR (34.8 KB)

    const int t       = threadIdx.x;
    const int lane    = t & 31;
    const int wave    = t >> 5;       // 0..7
    const int waveM   = wave & 3;     // 32-row slab
    const int waveN   = wave >> 2;    // 64-col slab
    const int ln15    = lane & 15;
    const int lh      = lane >> 4;
    const int rowBase = blockIdx.x * BM;
    const int cb      = blockIdx.y;
    const int colBase = cb * BN;

    v8f acc[2][4];
    #pragma unroll
    for (int i = 0; i < 2; ++i)
        #pragma unroll
        for (int j = 0; j < 4; ++j)
            #pragma unroll
            for (int v = 0; v < 8; ++v) acc[i][j][v] = 0.0f;

    for (int kt = 0; kt < KT; ++kt) {
        const int kBase = kt * BK;
        // ---- x tile: BM x BK fp32 -> bf16 LDS (coalesced: 16 lanes per 256B row span)
        #pragma unroll
        for (int i = 0; i < 8; ++i) {
            int idx = i * 256 + t;       // float4 index 0..2047
            int row = idx >> 4;
            int c4  = idx & 15;
            const float4 f = *(const float4*)(x + (size_t)(rowBase + row) * DIM + kBase + c4 * 4);
            uint2 p; p.x = pk_bf16(f.x, f.y); p.y = pk_bf16(f.z, f.w);
            *(uint2*)(As + row * ASTR + c4 * 4) = p;
        }
        // ---- W tile: BN x BK fp32 -> bf16 LDS (rows clamped on the ragged last block)
        #pragma unroll
        for (int i = 0; i < 8; ++i) {
            int idx = i * 256 + t;
            int row = idx >> 4;
            int c4  = idx & 15;
            int cls = colBase + row; cls = (cls > NCLS - 1) ? (NCLS - 1) : cls;
            const float4 f = *(const float4*)(W + (size_t)cls * DIM + kBase + c4 * 4);
            uint2 p; p.x = pk_bf16(f.x, f.y); p.y = pk_bf16(f.z, f.w);
            *(uint2*)(Bs + row * BSTR + c4 * 4) = p;
        }
        __syncthreads();

        #pragma unroll
        for (int kc = 0; kc < 2; ++kc) {
            const int k0 = kc * 32;
            v16bf a[2], b[4];
            #pragma unroll
            for (int tm = 0; tm < 2; ++tm) {      // A 16x32: K kb..kb+7 then kb+16..kb+23
                const __bf16* p = As + (waveM * 32 + tm * 16 + ln15) * ASTR + k0 + lh * 8;
                FragU u;
                u.q[0] = *(const uint4*)(p);
                u.q[1] = *(const uint4*)(p + 16);
                a[tm] = u.v;
            }
            #pragma unroll
            for (int tn = 0; tn < 4; ++tn) {      // B 32x16: lane=col, K kb..kb+7 then kb+16..kb+23
                const __bf16* p = Bs + (waveN * 64 + tn * 16 + ln15) * BSTR + k0 + lh * 8;
                FragU u;
                u.q[0] = *(const uint4*)(p);
                u.q[1] = *(const uint4*)(p + 16);
                b[tn] = u.v;
            }
            #pragma unroll
            for (int tm = 0; tm < 2; ++tm)
                #pragma unroll
                for (int tn = 0; tn < 4; ++tn)
                    acc[tm][tn] = __builtin_amdgcn_wmma_f32_16x16x32_bf16(
                        false, a[tm], false, b[tn], (short)0, acc[tm][tn], false, false);
        }
        __syncthreads();
    }

    // ---- epilogue: spill logits to LDS in two 64-col halves, online logsumexp per row
    const int colsValid = (NCLS - colBase < BN) ? (NCLS - colBase) : BN;
    const int tg = (t < BM) ? targ[rowBase + t] : 0;

    float m = -3.0e38f, s = 0.0f, tl = 0.0f;
    bool haveT = false;

    #pragma unroll
    for (int half = 0; half < 2; ++half) {
        __syncthreads();
        if (waveN == half) {
            #pragma unroll
            for (int tm = 0; tm < 2; ++tm)
                #pragma unroll
                for (int tn = 0; tn < 4; ++tn) {
                    int row = waveM * 32 + tm * 16 + lh * 8;  // C layout: VGPR v -> row+v
                    int col = tn * 16 + ln15;
                    #pragma unroll
                    for (int v = 0; v < 8; ++v)
                        Ls[(row + v) * LSTR + col] = acc[tm][tn][v];
                }
        }
        __syncthreads();
        if (t < BM) {
            int c0 = half * 64;
            int cn = colsValid - c0; cn = (cn > 64) ? 64 : cn;
            for (int c = 0; c < cn; ++c) {
                float vv = Ls[t * LSTR + c];
                if (vv > m) { s = s * __expf(m - vv) + 1.0f; m = vv; }
                else        { s += __expf(vv - m); }
            }
            int tloc = tg - colBase - c0;
            if (tloc >= 0 && tloc < cn) { tl = Ls[t * LSTR + tloc]; haveT = true; }
        }
    }
    if (t < BM) {
        Mpart[(size_t)cb * NTOK + rowBase + t] = m;
        Spart[(size_t)cb * NTOK + rowBase + t] = s;
        if (haveT) Tlog[rowBase + t] = tl;     // exactly one block owns each row's target
    }
}

// Stage 2: combine 393 (m,s) partials per row, mean NLL -> d_out[0].
__global__ __launch_bounds__(1024) void ce_reduce_kernel(
    const float* __restrict__ Mpart, const float* __restrict__ Spart,
    const float* __restrict__ Tlog, float* __restrict__ out)
{
    __shared__ float red[1024];
    const int t = threadIdx.x;
    float accl = 0.0f;
    for (int r = t; r < NTOK; r += 1024) {
        float M = -3.0e38f, S = 0.0f;
        for (int j = 0; j < CB; ++j) {
            float mj = Mpart[(size_t)j * NTOK + r];
            float sj = Spart[(size_t)j * NTOK + r];
            if (mj > M) { S = S * __expf(M - mj) + sj; M = mj; }
            else        { S += sj * __expf(mj - M); }
        }
        accl += (M + __logf(S)) - Tlog[r];
    }
    red[t] = accl;
    __syncthreads();
    for (int sft = 512; sft > 0; sft >>= 1) {
        if (t < sft) red[t] += red[t + sft];
        __syncthreads();
    }
    if (t == 0) out[0] = red[0] / (float)NTOK;
}

extern "C" void kernel_launch(void* const* d_in, const int* in_sizes, int n_in,
                              void* d_out, int out_size, void* d_ws, size_t ws_size,
                              hipStream_t stream) {
    (void)in_sizes; (void)n_in; (void)out_size; (void)ws_size;
    const float* x    = (const float*)d_in[0];
    const int*   targ = (const int*)d_in[1];
    const float* W    = (const float*)d_in[2];

    float* Mpart = (float*)d_ws;                       // CB*NTOK floats
    float* Spart = Mpart + (size_t)CB * NTOK;          // CB*NTOK floats
    float* Tlog  = Spart + (size_t)CB * NTOK;          // NTOK floats (~12.9 MB total)

    ce_partial_kernel<<<dim3(NTOK / BM, CB), 256, 0, stream>>>(x, targ, W, Mpart, Spart, Tlog);
    ce_reduce_kernel<<<1, 1024, 0, stream>>>(Mpart, Spart, Tlog, (float*)d_out);
}
